// SpikeDrivenBlock_33019708571665
// MI455X (gfx1250) — matrix operand order repaired and checked by reading, not compiled, not run on hardware
//
#include <hip/hip_runtime.h>

typedef __attribute__((ext_vector_type(16))) _Float16 v16h;
typedef __attribute__((ext_vector_type(8)))  float    v8f;

#define TT 4
#define BB 32
#define CC 384
#define CHID 1536
#define HW 196
#define HWP 208          // 13 tiles of 16
#define NTILES 13
#define LROW 136         // 128 K-chunk halves + 8 pad (avoids LDS bank conflicts)

// ---------------- Weight pre-swizzle: fp32 [COUT][KDIM] -> fp16 A-fragment order ----------------
// Fragment layout per (cout-tile ot, k-step ks): 32 lanes x 16 halves contiguous (1024 B).
// lane L<16 -> row ot*16+L, halves 0..7 = K ksBase..+7, halves 8..15 = K ksBase+16..+23
// lane L>=16 -> row ot*16+L-16, same with K offset +8.
template<int KDIM, int COUT>
__global__ __launch_bounds__(256) void swizzle_w_kernel(const float* __restrict__ W,
                                                        _Float16* __restrict__ Wf) {
  int d = blockIdx.x * 256 + threadIdx.x;      // over COUT*KDIM
  if (d >= COUT * KDIM) return;
  int i    = d & 15;
  int lane = (d >> 4) & 31;
  int rest = d >> 9;                            // ot * (KDIM/32) + ksIdx
  int ksIdx = rest % (KDIM / 32);
  int ot    = rest / (KDIM / 32);
  int row = ot * 16 + (lane & 15);
  int col = ksIdx * 32 + ((lane >> 4) << 3) + i + (i >= 8 ? 8 : 0);
  Wf[d] = (_Float16)W[(size_t)row * KDIM + col];
}

// B fragment (32x16 f16) from LDS tile stored [t][n][k] (row LROW halves):
// lane<16: col n=lane, halves 0..7 -> K = kbase+i, halves 8..15 -> K = kbase+16+(i-8) ; lane>=16: col n=lane-16, same with K offset +8
__device__ __forceinline__ v16h load_b_frag(const _Float16* __restrict__ lds, int t, int kbase, int lane) {
  int n  = lane & 15;
  int kh = kbase + ((lane >> 4) << 3);
  const _Float16* p = lds + (t * 16 + n) * LROW + kh;
  union { v16h v; uint4 u[2]; } r;
  r.u[0] = *reinterpret_cast<const uint4*>(p);
  r.u[1] = *reinterpret_cast<const uint4*>(p + 16);
  return r.v;
}

// ---------------- GEMM core: 4 timesteps share one A fragment ----------------
// Xs: spike activations [T][BB][KDIM][HWP] fp16 (pad columns zero)
// Wf: pre-swizzled fp16 weights in A-fragment order
// USEKV: multiply staged activations by per-(t,b,c) kv spike (a = q * kv) during staging
template<int KDIM, bool USEKV>
__device__ __forceinline__ void gemm4t(const _Float16* __restrict__ Xs,
                                       const _Float16* __restrict__ kvs,
                                       const _Float16* __restrict__ Wf,
                                       int b, int n0, int coutBase,
                                       _Float16* lds, v8f acc[TT]) {
  const int tid  = threadIdx.x;
  const int lane = tid & 31;
  const int ot   = coutBase >> 4;
  for (int kc = 0; kc < KDIM / 128; ++kc) {
    __syncthreads();
    // cooperative stage of 4t x 128k x 16n spikes, transposed to [t][n][k]
    #pragma unroll
    for (int i = 0; i < 4; ++i) {
      int id = tid + 256 * i;                   // 0..1023 = t(4) x k(128) x g(2)
      int g = id & 1, k = (id >> 1) & 127, t = id >> 8;
      int c = kc * 128 + k;
      const _Float16* src = Xs + ((size_t)((t * BB + b) * KDIM + c)) * HWP + n0 + g * 8;
      union { uint4 u; _Float16 h[8]; } d;
      d.u = *reinterpret_cast<const uint4*>(src);
      if (USEKV) {
        _Float16 m = kvs[(t * BB + b) * KDIM + c];
        #pragma unroll
        for (int j = 0; j < 8; ++j) d.h[j] *= m;
      }
      #pragma unroll
      for (int j = 0; j < 8; ++j)
        lds[(t * 16 + g * 8 + j) * LROW + k] = d.h[j];
    }
    __syncthreads();
    #pragma unroll
    for (int ks = 0; ks < 4; ++ks) {
      // A fragment: two coalesced b128 loads from pre-swizzled weights
      const _Float16* fb = Wf + ((size_t)(ot * (KDIM / 32) + kc * 4 + ks) << 9) + lane * 16;
      union { v16h v; uint4 u[2]; } a;
      a.u[0] = *reinterpret_cast<const uint4*>(fb);
      a.u[1] = *reinterpret_cast<const uint4*>(fb + 8);
      // preload all 4 timestep B fragments, then issue WMMAs back-to-back
      v16h bf[TT];
      #pragma unroll
      for (int t = 0; t < TT; ++t) bf[t] = load_b_frag(lds, t, ks * 32, lane);
      #pragma unroll
      for (int t = 0; t < TT; ++t)
        acc[t] = __builtin_amdgcn_wmma_f32_16x16x32_f16(
            false, a.v, false, bf[t], (short)0, acc[t], false, false);
    }
  }
}

// ---------------- Kernel 1: LIF on raw input, fp32 -> padded fp16 spikes ----------------
__global__ __launch_bounds__(256) void lif_in_kernel(const float* __restrict__ x,
                                                     _Float16* __restrict__ xs) {
  int idx = blockIdx.x * 256 + threadIdx.x;     // over BB*CC*HWP
  if (idx >= BB * CC * HWP) return;
  int n = idx % HWP;
  int bc = idx / HWP;
  bool valid = n < HW;
  float memb = 0.f;
  #pragma unroll
  for (int t = 0; t < TT; ++t) {
    float xv = valid ? x[(size_t)(t * BB * CC + bc) * HW + n] : 0.f;
    memb = 0.5f * (memb + xv);                  // v += (x - v)/tau, tau=2
    float s = (memb >= 1.0f) ? 1.f : 0.f;
    xs[(size_t)(t * BB * CC + bc) * HWP + n] = (_Float16)(valid ? s : 0.f);
    memb = (s > 0.f) ? 0.f : memb;
  }
}

// ---------------- Kernel 2: GEMM + BN + LIF -> spikes (used for q,k,v,fc1) ----------------
template<int KDIM, int COUT>
__global__ __launch_bounds__(256) void gemm_bn_lif_kernel(const _Float16* __restrict__ Xs,
                                                          const _Float16* __restrict__ Wf,
                                                          const float* __restrict__ bn,
                                                          const float* __restrict__ bias,
                                                          _Float16* __restrict__ Sout) {
  __shared__ _Float16 lds[TT * 16 * LROW];
  int bx = blockIdx.x;
  int b = bx / NTILES, n0 = (bx % NTILES) * 16;
  int wave = threadIdx.x >> 5, lane = threadIdx.x & 31;
  int coutBase = blockIdx.y * 128 + wave * 16;
  v8f acc[TT];
  #pragma unroll
  for (int t = 0; t < TT; ++t)
    #pragma unroll
    for (int r = 0; r < 8; ++r) acc[t][r] = 0.f;
  gemm4t<KDIM, false>(Xs, nullptr, Wf, b, n0, coutBase, lds, acc);

  int nl = lane & 15, hi = lane >> 4;
  int hw = n0 + nl;
  bool valid = hw < HW;
  #pragma unroll
  for (int r = 0; r < 8; ++r) {
    int o = coutBase + r + hi * 8;
    float g = bn[o], be = bn[COUT + o], mm = bn[2 * COUT + o], vv = bn[3 * COUT + o];
    float sc = g * rsqrtf(vv + 1e-5f);
    float sh = be - mm * sc;
    if (bias) sh += bias[o] * sc;
    float memb = 0.f;
    #pragma unroll
    for (int t = 0; t < TT; ++t) {
      float y = acc[t][r] * sc + sh;
      memb = 0.5f * (memb + y);
      float s = (memb >= 1.0f) ? 1.f : 0.f;
      Sout[(size_t)((t * BB + b) * COUT + o) * HWP + hw] = (_Float16)(valid ? s : 0.f);
      memb = (s > 0.f) ? 0.f : memb;
    }
  }
}

// ---------------- Kernel 3: kv = sum_n k*v per (b,c), LIF(vth=0.5) -> kvs spikes ----------------
__global__ __launch_bounds__(256) void kv_lif_kernel(const _Float16* __restrict__ Ks,
                                                     const _Float16* __restrict__ Vs,
                                                     _Float16* __restrict__ kvs) {
  int idx = blockIdx.x * 256 + threadIdx.x;    // over BB*CC
  if (idx >= BB * CC) return;
  float memb = 0.f;
  #pragma unroll
  for (int t = 0; t < TT; ++t) {
    const uint4* kp = reinterpret_cast<const uint4*>(Ks + (size_t)(t * BB * CC + idx) * HWP);
    const uint4* vp = reinterpret_cast<const uint4*>(Vs + (size_t)(t * BB * CC + idx) * HWP);
    float sum = 0.f;
    for (int i = 0; i < HWP / 8; ++i) {       // pad columns are zero -> safe
      union { uint4 u; _Float16 h[8]; } a, c;
      a.u = kp[i]; c.u = vp[i];
      #pragma unroll
      for (int j = 0; j < 8; ++j) sum += (float)a.h[j] * (float)c.h[j];
    }
    memb = 0.5f * (memb + sum);
    float s = (memb >= 0.5f) ? 1.f : 0.f;
    kvs[t * BB * CC + idx] = (_Float16)s;
    memb = (s > 0.f) ? 0.f : memb;
  }
}

// ---------------- Kernel 4: proj GEMM (+bias,BN,+x) -> out1 (f32) and LIF spikes s1 ----------------
__global__ __launch_bounds__(256) void proj_kernel(const _Float16* __restrict__ Qs,
                                                   const _Float16* __restrict__ kvs,
                                                   const _Float16* __restrict__ Wf,
                                                   const float* __restrict__ bn,
                                                   const float* __restrict__ bias,
                                                   const float* __restrict__ x,
                                                   float* __restrict__ out1,
                                                   _Float16* __restrict__ s1) {
  __shared__ _Float16 lds[TT * 16 * LROW];
  int bx = blockIdx.x;
  int b = bx / NTILES, n0 = (bx % NTILES) * 16;
  int wave = threadIdx.x >> 5, lane = threadIdx.x & 31;
  int coutBase = blockIdx.y * 128 + wave * 16;
  v8f acc[TT];
  #pragma unroll
  for (int t = 0; t < TT; ++t)
    #pragma unroll
    for (int r = 0; r < 8; ++r) acc[t][r] = 0.f;
  gemm4t<CC, true>(Qs, kvs, Wf, b, n0, coutBase, lds, acc);  // B = q * kv applied in staging

  int nl = lane & 15, hi = lane >> 4;
  int hw = n0 + nl;
  bool valid = hw < HW;
  #pragma unroll
  for (int r = 0; r < 8; ++r) {
    int o = coutBase + r + hi * 8;
    float g = bn[o], be = bn[CC + o], mm = bn[2 * CC + o], vv = bn[3 * CC + o];
    float sc = g * rsqrtf(vv + 1e-5f);
    float sh = (bias[o] - mm) * sc + be;
    float memb = 0.f;
    #pragma unroll
    for (int t = 0; t < TT; ++t) {
      size_t pidx = (size_t)((t * BB + b) * CC + o) * HWP + hw;
      float idv = valid ? x[(size_t)((t * BB + b) * CC + o) * HW + hw] : 0.f;
      float o1 = valid ? (acc[t][r] * sc + sh + idv) : 0.f;
      out1[pidx] = o1;
      memb = 0.5f * (memb + o1);
      float s = (memb >= 1.0f) ? 1.f : 0.f;
      s1[pidx] = (_Float16)(valid ? s : 0.f);
      memb = (s > 0.f) ? 0.f : memb;
    }
  }
}

// ---------------- Kernel 6: fc2 GEMM (+bias,BN,+out1) -> final output (f32, unpadded) --------
__global__ __launch_bounds__(256) void fc2_kernel(const _Float16* __restrict__ S2,
                                                  const _Float16* __restrict__ Wf,
                                                  const float* __restrict__ bn,
                                                  const float* __restrict__ bias,
                                                  const float* __restrict__ out1,
                                                  float* __restrict__ out) {
  __shared__ _Float16 lds[TT * 16 * LROW];
  int bx = blockIdx.x;
  int b = bx / NTILES, n0 = (bx % NTILES) * 16;
  int wave = threadIdx.x >> 5, lane = threadIdx.x & 31;
  int coutBase = blockIdx.y * 128 + wave * 16;
  v8f acc[TT];
  #pragma unroll
  for (int t = 0; t < TT; ++t)
    #pragma unroll
    for (int r = 0; r < 8; ++r) acc[t][r] = 0.f;
  gemm4t<CHID, false>(S2, nullptr, Wf, b, n0, coutBase, lds, acc);

  int nl = lane & 15, hi = lane >> 4;
  int hw = n0 + nl;
  bool valid = hw < HW;
  #pragma unroll
  for (int r = 0; r < 8; ++r) {
    int o = coutBase + r + hi * 8;
    float g = bn[o], be = bn[CC + o], mm = bn[2 * CC + o], vv = bn[3 * CC + o];
    float sc = g * rsqrtf(vv + 1e-5f);
    float sh = (bias[o] - mm) * sc + be;
    #pragma unroll
    for (int t = 0; t < TT; ++t) {
      float y = acc[t][r] * sc + sh +
                out1[(size_t)((t * BB + b) * CC + o) * HWP + hw];
      if (valid) out[(size_t)((t * BB + b) * CC + o) * HW + hw] = y;
    }
  }
}

// ---------------- launch ----------------
extern "C" void kernel_launch(void* const* d_in, const int* in_sizes, int n_in,
                              void* d_out, int out_size, void* d_ws, size_t ws_size,
                              hipStream_t stream) {
  (void)in_sizes; (void)n_in; (void)out_size; (void)ws_size;
  const float* x      = (const float*)d_in[0];
  const float* q_w    = (const float*)d_in[1];
  const float* k_w    = (const float*)d_in[2];
  const float* v_w    = (const float*)d_in[3];
  const float* q_bn   = (const float*)d_in[4];
  const float* k_bn   = (const float*)d_in[5];
  const float* v_bn   = (const float*)d_in[6];
  const float* proj_w = (const float*)d_in[7];
  const float* proj_b = (const float*)d_in[8];
  const float* proj_bn= (const float*)d_in[9];
  const float* fc1_w  = (const float*)d_in[10];
  const float* fc1_b  = (const float*)d_in[11];
  const float* fc1_bn = (const float*)d_in[12];
  const float* fc2_w  = (const float*)d_in[13];
  const float* fc2_b  = (const float*)d_in[14];
  const float* fc2_bn = (const float*)d_in[15];

  char* ws = (char*)d_ws;
  const size_t SP16 = (size_t)TT * BB * CC * HWP * sizeof(_Float16);  // 20.45 MB
  const size_t WSML = (size_t)CC * CC * sizeof(_Float16);             // 294,912 B
  const size_t WBIG = (size_t)CHID * CC * sizeof(_Float16);           // 1,179,648 B
  _Float16* xs   = (_Float16*)(ws);
  _Float16* qs   = (_Float16*)(ws + 1 * SP16);
  _Float16* ks   = (_Float16*)(ws + 2 * SP16);
  _Float16* vs   = (_Float16*)(ws + 3 * SP16);
  _Float16* kvsb = (_Float16*)(ws + 4 * SP16);                         // 96 KB used
  float*    out1 = (float*)   (ws + 4 * SP16 + 131072);                // 2*SP16 bytes
  _Float16* s2   = (_Float16*)(ws + 6 * SP16 + 131072);                // 4*SP16 bytes
  char*     wbase= ws + 10 * SP16 + 131072;
  _Float16* qf   = (_Float16*)(wbase);
  _Float16* kf   = (_Float16*)(wbase + 1 * WSML);
  _Float16* vf   = (_Float16*)(wbase + 2 * WSML);
  _Float16* pf   = (_Float16*)(wbase + 3 * WSML);
  _Float16* f1f  = (_Float16*)(wbase + 4 * WSML);
  _Float16* f2f  = (_Float16*)(wbase + 4 * WSML + WBIG);
  _Float16* s1   = ks;  // ks fully consumed by kv_lif before proj writes s1

  // 0) pre-swizzle all weights into WMMA A-fragment order (fp16)
  int gS = (CC * CC + 255) / 256, gB = (CHID * CC + 255) / 256;
  swizzle_w_kernel<CC, CC>    <<<gS, 256, 0, stream>>>(q_w,    qf);
  swizzle_w_kernel<CC, CC>    <<<gS, 256, 0, stream>>>(k_w,    kf);
  swizzle_w_kernel<CC, CC>    <<<gS, 256, 0, stream>>>(v_w,    vf);
  swizzle_w_kernel<CC, CC>    <<<gS, 256, 0, stream>>>(proj_w, pf);
  swizzle_w_kernel<CC, CHID>  <<<gB, 256, 0, stream>>>(fc1_w,  f1f);
  swizzle_w_kernel<CHID, CC>  <<<gB, 256, 0, stream>>>(fc2_w,  f2f);

  // 1) input LIF
  lif_in_kernel<<<(BB * CC * HWP + 255) / 256, 256, 0, stream>>>(x, xs);

  // 2) q/k/v: GEMM + BN + LIF
  dim3 gq(BB * NTILES, CC / 128);
  gemm_bn_lif_kernel<CC, CC><<<gq, 256, 0, stream>>>(xs, qf, q_bn, nullptr, qs);
  gemm_bn_lif_kernel<CC, CC><<<gq, 256, 0, stream>>>(xs, kf, k_bn, nullptr, ks);
  gemm_bn_lif_kernel<CC, CC><<<gq, 256, 0, stream>>>(xs, vf, v_bn, nullptr, vs);

  // 3) kv reduction + LIF(0.5)
  kv_lif_kernel<<<(BB * CC + 255) / 256, 256, 0, stream>>>(ks, vs, kvsb);

  // 4) proj: (q*kv) GEMM + bias + BN + residual -> out1, LIF -> s1
  proj_kernel<<<gq, 256, 0, stream>>>(qs, kvsb, pf, proj_bn, proj_b, x, out1, s1);

  // 5) fc1: GEMM + bias + BN + LIF -> s2
  dim3 g1(BB * NTILES, CHID / 128);
  gemm_bn_lif_kernel<CC, CHID><<<g1, 256, 0, stream>>>(s1, f1f, fc1_bn, fc1_b, s2);

  // 6) fc2: GEMM + bias + BN + residual -> out
  fc2_kernel<<<gq, 256, 0, stream>>>(s2, f2f, fc2_bn, fc2_b, out1, (float*)d_out);
}
